// GridRNN_10539849744974
// MI455X (gfx1250) — matrix operand order repaired and checked by reading, not compiled, not run on hardware
//
#include <hip/hip_runtime.h>
#include <math.h>

typedef __bf16 v16bf __attribute__((ext_vector_type(16)));
typedef float  v8f   __attribute__((ext_vector_type(8)));

#define BB  4
#define SS  48
#define TT  48
#define HD  512
#define DEP 3
#define NT16 (HD/16)        /* 32 N-tiles of 16 */
#define KT32 (HD/32)        /* 16 K-tiles of 32 */
#define WP_KT_STRIDE (NT16*32*16)  /* elements per kt plane in packed weights */
#define ROWS 48             /* rows per batch in both recurrences (S==T) */
#define HALF_BLOCKS 48      /* 384 tiles / 8 waves per block */

__device__ __forceinline__ __bf16 f2bf(float f) {
  unsigned u = __builtin_bit_cast(unsigned, f);
  u += 0x7FFFu + ((u >> 16) & 1u);               // round-to-nearest-even
  unsigned short s = (unsigned short)(u >> 16);
  return __builtin_bit_cast(__bf16, s);
}

// A fragment (16x32 bf16) per CDNA5 layout: lane<16 -> M=lane, K {kb..kb+7, kb+16..kb+23}
__device__ __forceinline__ v16bf load_a_bf16(const __bf16* __restrict__ ap) {
  v16bf a;
#pragma unroll
  for (int e = 0; e < 8; ++e) { a[e] = ap[e]; a[8 + e] = ap[16 + e]; }
  return a;
}
__device__ __forceinline__ v16bf load_a_f32(const float* __restrict__ ap) {
  v16bf a;
#pragma unroll
  for (int e = 0; e < 8; ++e) { a[e] = f2bf(ap[e]); a[8 + e] = f2bf(ap[16 + e]); }
  return a;
}

// ---- pack fp32 W[K=512][N=512] into WMMA B-fragment order (bf16) ----
// layout: [kt][nt][lane 0..31][e 0..15]; N=lane&15, K=kt*32+(lane>>4)*8+(e&7)+(e>>3)*16
__global__ void pack_weight(const float* __restrict__ W, __bf16* __restrict__ Wp) {
  size_t idx = (size_t)blockIdx.x * blockDim.x + threadIdx.x;
  if (idx >= (size_t)HD * HD) return;
  int e    = idx & 15;
  int lane = (idx >> 4) & 31;
  int t    = (int)(idx >> 9);
  int nt   = t & (NT16 - 1);
  int kt   = t >> 5;
  int k = kt * 32 + (lane >> 4) * 8 + (e & 7) + (e >> 3) * 16;
  int n = nt * 16 + (lane & 15);
  Wp[idx] = f2bf(W[(size_t)k * HD + n]);
}

__global__ void zero_u32(unsigned* __restrict__ p, int n) {
  int i = blockIdx.x * blockDim.x + threadIdx.x;
  if (i < n) p[i] = 0u;
}

// build depth-d input (interleave of prev depth hx/hy) as bf16 [B,S,T,H]
__global__ void build_inter(const float* __restrict__ prevBase, __bf16* __restrict__ inter) {
  size_t idx = (size_t)blockIdx.x * blockDim.x + threadIdx.x;
  const size_t total = (size_t)BB * SS * TT * HD;
  if (idx >= total) return;
  int h = idx & (HD - 1);
  size_t r = idx >> 9;
  int j = (int)(r % TT); r /= TT;
  int i = (int)(r % SS); r /= SS;
  int b = (int)r;
  const float* src = prevBase
      + (size_t)b * ((size_t)DEP * SS * TT * 2 * HD)
      + ((size_t)i * (TT * 2) + (size_t)(j >> 1) * 2 + (j & 1)) * HD + h;
  inter[idx] = f2bf(*src);
}

// ---- generic WMMA GEMM:  C[M,512] = A[M,512] * Wpack + bias  (fp32 out) ----
template <bool ABF16>
__global__ __launch_bounds__(256) void wmma_gemm_bias(
    const void* __restrict__ Aptr, const __bf16* __restrict__ Wp,
    const float* __restrict__ bias, float* __restrict__ C, int M) {
  const int lane = threadIdx.x & 31;
  const int wave = threadIdx.x >> 5;
  int tile = blockIdx.x * 8 + wave;
  int tile_m = tile / NT16, tile_n = tile % NT16;
  if (tile_m * 16 >= M) return;                  // wave-uniform
  const int mrow = tile_m * 16 + (lane & 15);
  const int asel = lane >> 4;                    // K half select
  const __bf16* ab = (const __bf16*)Aptr + (size_t)mrow * HD + asel * 8;
  const float*  af = (const float*)Aptr + (size_t)mrow * HD + asel * 8;
  const __bf16* wp = Wp + ((size_t)tile_n * 32 + lane) * 16;
  v8f acc = {};
  // software-pipelined: fetch kt+1 before consuming kt
  v16bf a = ABF16 ? load_a_bf16(ab) : load_a_f32(af);
  v16bf b = *(const v16bf*)wp;
#pragma unroll 4
  for (int kt = 0; kt < KT32 - 1; ++kt) {
    ab += 32; af += 32; wp += WP_KT_STRIDE;
    v16bf an = ABF16 ? load_a_bf16(ab) : load_a_f32(af);
    v16bf bn = *(const v16bf*)wp;
    acc = __builtin_amdgcn_wmma_f32_16x16x32_bf16(false, a, false, b, (short)0, acc, false, false);
    a = an; b = bn;
  }
  acc = __builtin_amdgcn_wmma_f32_16x16x32_bf16(false, a, false, b, (short)0, acc, false, false);
  const int n = tile_n * 16 + (lane & 15);
  const float bv = bias[n];
#pragma unroll
  for (int r = 0; r < 8; ++r) {
    int m = tile_m * 16 + asel * 8 + r;
    C[(size_t)m * HD + n] = acc[r] + bv;
  }
}

// ---- fused recurrence step: X half (roll over j) + Y half in one launch ----
struct StepHalf {
  const __bf16* state_in;   // [192,512] bf16
  __bf16* state_out;        // [192,512] bf16
  const __bf16* Wp;         // packed Wh
  const float* aterm;       // pre-activation base for this step
  unsigned long long ab, ai;// aterm strides: per-b, per-inner (elements)
  float* outp;              // output base for this step (channel folded in)
  unsigned long long ob, oi;// out strides: per-b, per-inner (elements)
  int roll;                 // 1 for X (sx[j]=h[j-1 mod ROWS]), 0 for Y
};

__global__ __launch_bounds__(256) void wmma_step_dual(StepHalf X, StepHalf Y) {
  const bool isX = (blockIdx.x < HALF_BLOCKS);
  const StepHalf P = isX ? X : Y;
  const int bid  = isX ? (int)blockIdx.x : (int)blockIdx.x - HALF_BLOCKS;
  const int lane = threadIdx.x & 31;
  const int wave = threadIdx.x >> 5;
  const int tile = bid * 8 + wave;               // 384 tiles / half, exact
  const int tile_m = tile / NT16, tile_n = tile % NT16;
  const int mrow  = tile_m * 16 + (lane & 15);
  const int bidx  = mrow / ROWS;                 // constant divisor -> magic mul
  const int inner = mrow - bidx * ROWS;
  const int sinner = P.roll ? (inner + ROWS - 1) % ROWS : inner;
  const int asel = lane >> 4;
  const __bf16* ap = P.state_in + (size_t)(bidx * ROWS + sinner) * HD + asel * 8;
  const __bf16* wp = P.Wp + ((size_t)tile_n * 32 + lane) * 16;
  v8f acc = {};
  v16bf a = load_a_bf16(ap);
  v16bf b = *(const v16bf*)wp;
#pragma unroll 4
  for (int kt = 0; kt < KT32 - 1; ++kt) {
    ap += 32; wp += WP_KT_STRIDE;
    v16bf an = load_a_bf16(ap);
    v16bf bn = *(const v16bf*)wp;
    acc = __builtin_amdgcn_wmma_f32_16x16x32_bf16(false, a, false, b, (short)0, acc, false, false);
    a = an; b = bn;
  }
  acc = __builtin_amdgcn_wmma_f32_16x16x32_bf16(false, a, false, b, (short)0, acc, false, false);
  const int n = tile_n * 16 + (lane & 15);
  // 8-row output window is 8-aligned; ROWS(48) is a multiple of 8 ->
  // window never crosses a batch boundary: compute mb/mi once.
  const int m0 = tile_m * 16 + asel * 8;
  const int mb = m0 / ROWS;                      // constant divisor
  const int mi0 = m0 - mb * ROWS;
  const float* apre = P.aterm + (size_t)mb * P.ab + (size_t)mi0 * P.ai + n;
  float* orow = P.outp + (size_t)mb * P.ob + (size_t)mi0 * P.oi + n;
  __bf16* srow = P.state_out + (size_t)m0 * HD + n;
#pragma unroll
  for (int r = 0; r < 8; ++r) {
    float v = tanhf(acc[r] + apre[(size_t)r * P.ai]);
    orow[(size_t)r * P.oi] = v;
    srow[(size_t)r * HD] = f2bf(v);
  }
}

extern "C" void kernel_launch(void* const* d_in, const int* in_sizes, int n_in,
                              void* d_out, int out_size, void* d_ws, size_t ws_size,
                              hipStream_t stream) {
  (void)in_sizes; (void)n_in; (void)out_size; (void)ws_size;
  const float* src   = (const float*)d_in[0];
  const float* trg   = (const float*)d_in[1];
  const float* Wi_x2 = (const float*)d_in[5];
  const float* Wh_x2 = (const float*)d_in[6];
  const float* b_x2  = (const float*)d_in[7];
  const float* Wi_x3 = (const float*)d_in[8];
  const float* Wh_x3 = (const float*)d_in[9];
  const float* b_x3  = (const float*)d_in[10];
  const float* Wi_y2 = (const float*)d_in[14];
  const float* Wh_y2 = (const float*)d_in[15];
  const float* b_y2  = (const float*)d_in[16];
  const float* Wi_y3 = (const float*)d_in[17];
  const float* Wh_y3 = (const float*)d_in[18];
  const float* b_y3  = (const float*)d_in[19];
  float* out = (float*)d_out;

  char* ws = (char*)d_ws;
  size_t off = 0;
  auto alloc = [&](size_t bytes) -> void* {
    off = (off + 255) & ~(size_t)255;
    void* p = ws + off; off += bytes; return p;
  };

  const size_t WB = (size_t)HD * HD * sizeof(__bf16);
  // slots: [set3: Wi_x, Wh_x, Wi_y, Wh_y][set2: same]
  __bf16* WP[8];
  for (int i = 0; i < 8; ++i) WP[i] = (__bf16*)alloc(WB);
  __bf16* inter = (__bf16*)alloc((size_t)BB * SS * TT * HD * sizeof(__bf16));
  float*  Ax    = (float*)alloc((size_t)BB * SS * TT * HD * sizeof(float));
  float*  Ay    = (float*)alloc((size_t)BB * SS * TT * HD * sizeof(float));
  float*  Ax0   = (float*)alloc((size_t)BB * SS * HD * sizeof(float));
  float*  Ay0   = (float*)alloc((size_t)BB * TT * HD * sizeof(float));
  __bf16* stx[2], *sty[2];
  for (int i = 0; i < 2; ++i) stx[i] = (__bf16*)alloc((size_t)BB * TT * HD * sizeof(__bf16));
  for (int i = 0; i < 2; ++i) sty[i] = (__bf16*)alloc((size_t)BB * SS * HD * sizeof(__bf16));

  // pack the 8 weight matrices actually used (sets 3 and 2; set 1 is dead per reference bug)
  const float* wsrc[8] = {Wi_x3, Wh_x3, Wi_y3, Wh_y3, Wi_x2, Wh_x2, Wi_y2, Wh_y2};
  for (int i = 0; i < 8; ++i)
    pack_weight<<<(HD * HD) / 256, 256, 0, stream>>>(wsrc[i], WP[i]);

  const unsigned long long outB = (unsigned long long)DEP * SS * TT * 2 * HD; // per-batch out stride
  const int stElems = BB * TT * HD;            // == BB*SS*HD (S==T)

  for (int d = 0; d < DEP; ++d) {
    const int sel = (d == 1) ? 1 : 0;          // depths use cells (3,2,3)
    __bf16* WIx = WP[sel * 4 + 0]; __bf16* WHx = WP[sel * 4 + 1];
    __bf16* WIy = WP[sel * 4 + 2]; __bf16* WHy = WP[sel * 4 + 3];
    const float* bx = sel ? b_x2 : b_x3;
    const float* by = sel ? b_y2 : b_y3;

    if (d == 0) {
      // xin depends only on (b,i); yin only on (b,j): tiny [192,512]x[512,512] GEMMs
      wmma_gemm_bias<false><<<(BB * SS / 16) * NT16 / 8, 256, 0, stream>>>(src, WIx, bx, Ax0, BB * SS);
      wmma_gemm_bias<false><<<(BB * TT / 16) * NT16 / 8, 256, 0, stream>>>(trg, WIy, by, Ay0, BB * TT);
    } else {
      const size_t tot = (size_t)BB * SS * TT * HD;
      build_inter<<<(unsigned)((tot + 255) / 256), 256, 0, stream>>>(
          out + (size_t)(d - 1) * SS * TT * 2 * HD, inter);
      const int M = BB * SS * TT;
      wmma_gemm_bias<true><<<(M / 16) * NT16 / 8, 256, 0, stream>>>(inter, WIx, bx, Ax, M);
      wmma_gemm_bias<true><<<(M / 16) * NT16 / 8, 256, 0, stream>>>(inter, WIy, by, Ay, M);
    }

    zero_u32<<<(stElems / 2 + 255) / 256, 256, 0, stream>>>((unsigned*)stx[0], stElems / 2);
    zero_u32<<<(stElems / 2 + 255) / 256, 256, 0, stream>>>((unsigned*)sty[0], stElems / 2);

    for (int s = 0; s < SS; ++s) {             // SS == TT == 48 fused steps
      StepHalf X, Y;
      X.state_in = stx[s & 1]; X.state_out = stx[(s & 1) ^ 1]; X.Wp = WHx;
      if (d == 0) { X.aterm = Ax0 + (size_t)s * HD; X.ab = (unsigned long long)SS * HD; X.ai = 0; }
      else        { X.aterm = Ax + (size_t)s * TT * HD; X.ab = (unsigned long long)SS * TT * HD; X.ai = HD; }
      X.outp = out + ((size_t)d * SS * TT * 2 + (size_t)s * TT * 2) * HD;
      X.ob = outB; X.oi = 2 * HD; X.roll = 1;

      Y.state_in = sty[s & 1]; Y.state_out = sty[(s & 1) ^ 1]; Y.Wp = WHy;
      if (d == 0) { Y.aterm = Ay0 + (size_t)s * HD; Y.ab = (unsigned long long)TT * HD; Y.ai = 0; }
      else        { Y.aterm = Ay + (size_t)s * HD; Y.ab = (unsigned long long)SS * TT * HD; Y.ai = (unsigned long long)TT * HD; }
      Y.outp = out + ((size_t)d * SS * TT * 2 + (size_t)s * 2) * HD + HD;
      Y.ob = outB; Y.oi = (unsigned long long)TT * 2 * HD; Y.roll = 0;

      wmma_step_dual<<<2 * HALF_BLOCKS, 256, 0, stream>>>(X, Y);
    }
  }
}
